// MultiHeadSelfAttention_47708496724663
// MI455X (gfx1250) — compile-verified, matrix operand order checked by reading, not run on hardware
//
#include <hip/hip_runtime.h>

#define S_LEN 2048
#define HID   1024
#define NH_   16
#define HD_   64
#define BATCH 2
#define NEGINF (-1.0e38f)

typedef __attribute__((ext_vector_type(16))) __bf16 v16bf;
typedef __attribute__((ext_vector_type(8)))  float  v8f;
typedef __attribute__((ext_vector_type(8)))  unsigned short us8;
typedef __attribute__((ext_vector_type(4)))  float  f4;

union Vb16 { v16bf v; us8 h[2]; unsigned short u[16]; };

__device__ __forceinline__ unsigned short f2bfbits(float f) {
  unsigned u = __float_as_uint(f);
  u += 0x7FFFu + ((u >> 16) & 1u);           // round-to-nearest-even
  return (unsigned short)(u >> 16);
}

// 16-bit A/B fragment K pattern (wave32, 16x16x32): per lane two contiguous
// 8-element runs at [kgrp, kgrp+8) and [16+kgrp, 16+kgrp+8), kgrp = 8*(lane>=16).
__device__ __forceinline__ v16bf load_frag_bf16(const unsigned short* rowp, int kgrp) {
  Vb16 u;
  u.h[0] = *(const us8*)(rowp + kgrp);
  u.h[1] = *(const us8*)(rowp + 16 + kgrp);
  return u.v;
}

__device__ __forceinline__ v8f wmma_bf16(v16bf a, v16bf b, v8f c) {
  return __builtin_amdgcn_wmma_f32_16x16x32_bf16(false, a, false, b, (short)0, c, false, false);
}

// reductions across the 16 lanes sharing a C-fragment row (xor<16 stays in each half-wave)
__device__ __forceinline__ float rmax16(float v) {
#pragma unroll
  for (int m = 1; m < 16; m <<= 1) v = fmaxf(v, __shfl_xor(v, m, 32));
  return v;
}
__device__ __forceinline__ float rsum16(float v) {
#pragma unroll
  for (int m = 1; m < 16; m <<= 1) v += __shfl_xor(v, m, 32);
  return v;
}

// ---------------------------------------------------------------------------
// Kernel 0: zero the descriptor accumulators
// ---------------------------------------------------------------------------
__global__ void init_acc_kernel(float* acc) {
  if (threadIdx.x < NH_ * 4) acc[threadIdx.x] = 0.0f;
}

// ---------------------------------------------------------------------------
// Pack kernels: x -> bf16 flat; W (KxN f32, row-major) -> bf16 NxK (transposed)
// ---------------------------------------------------------------------------
__global__ void __launch_bounds__(256) pack_x_kernel(const float* __restrict__ x,
                                                     unsigned short* __restrict__ xb) {
  size_t base = ((size_t)blockIdx.x * 256 + threadIdx.x) * 8;
  f4 a = *(const f4*)(x + base);
  f4 b = *(const f4*)(x + base + 4);
  us8 o;
#pragma unroll
  for (int j = 0; j < 4; j++) { o[j] = f2bfbits(a[j]); o[4 + j] = f2bfbits(b[j]); }
  *(us8*)(xb + base) = o;
}

__global__ void __launch_bounds__(256) packT_kernel(const float* __restrict__ W,
                                                    unsigned short* __restrict__ Wt,
                                                    int K, int N) {
  __shared__ float tile[32][33];
  const int n0 = blockIdx.x * 32, k0 = blockIdx.y * 32;
  const int tx = threadIdx.x & 31, ty = threadIdx.x >> 5;   // 32 x 8
#pragma unroll
  for (int r = 0; r < 32; r += 8)
    tile[ty + r][tx] = W[(size_t)(k0 + ty + r) * N + n0 + tx];
  __syncthreads();
#pragma unroll
  for (int r = 0; r < 32; r += 8)
    Wt[(size_t)(n0 + ty + r) * K + k0 + tx] = f2bfbits(tile[tx][ty + r]);
}

// ---------------------------------------------------------------------------
// Kernel 1: qkv = x @ Wqkv + bqkv  (M=4096, N=3072, K=1024), bf16 WMMA / f32 acc
// Q,K -> [B,NH,S,64] bf16 ; V -> transposed [B,NH,64,S] bf16
// ---------------------------------------------------------------------------
__global__ void __launch_bounds__(256) qkv_gemm_kernel(
    const unsigned short* __restrict__ xb, const unsigned short* __restrict__ WqkvT,
    const float* __restrict__ bqkv,
    unsigned short* __restrict__ Qb, unsigned short* __restrict__ Kb,
    unsigned short* __restrict__ Vt) {
  const int lane = threadIdx.x & 31;
  const int wave = threadIdx.x >> 5;
  const int tile = blockIdx.x * 8 + wave;          // 256*192 tiles total
  const int NT_N = (3 * HID) / 16;                 // 192
  const int tm = tile / NT_N, tn = tile % NT_N;
  const int col = lane & 15;
  const int grp = (lane & 16) ? 8 : 0;
  const int kgrp = grp;

  v8f acc;
#pragma unroll
  for (int i = 0; i < 8; i++) acc[i] = 0.0f;

  const unsigned short* xrow = xb + (size_t)(tm * 16 + col) * HID;
  const unsigned short* wrow = WqkvT + (size_t)(tn * 16 + col) * HID;
  for (int kk = 0; kk < HID; kk += 32) {
    if (kk + 32 < HID) {
      __builtin_prefetch(xrow + kk + 32, 0, 1);
      __builtin_prefetch(wrow + kk + 32, 0, 1);
    }
    v16bf ua = load_frag_bf16(xrow + kk, kgrp);
    v16bf ub = load_frag_bf16(wrow + kk, kgrp);
    acc = wmma_bf16(ua, ub, acc);
  }

  const int n = tn * 16 + col;
  const float bias = bqkv[n];
  const int t = n / HID, rem = n % HID;
  const int hh = rem / HD_, d = rem % HD_;
  if (t == 2) {                                    // V: transposed, vector store
    us8 vv;
#pragma unroll
    for (int i = 0; i < 8; i++) vv[i] = f2bfbits(acc[i] + bias);
    int m0 = tm * 16 + grp;
    int bb = m0 / S_LEN, sq0 = m0 % S_LEN;         // 16-row tile never straddles batch
    *(us8*)&Vt[((size_t)(bb * NH_ + hh) * HD_ + d) * S_LEN + sq0] = vv;
  } else {
    unsigned short* dst = (t == 0) ? Qb : Kb;
#pragma unroll
    for (int i = 0; i < 8; i++) {
      int m = tm * 16 + i + grp;
      int bb = m / S_LEN, sq = m % S_LEN;
      dst[(((size_t)(bb * NH_ + hh) * S_LEN + sq) * HD_) + d] = f2bfbits(acc[i] + bias);
    }
  }
}

// ---------------------------------------------------------------------------
// Kernel 2: fused causal attention + online descriptors.
// One wave per 16-query tile; block = 8 waves covering 128 queries of one (b,h).
// ---------------------------------------------------------------------------
__global__ void __launch_bounds__(256) attn_kernel(
    const unsigned short* __restrict__ Qb, const unsigned short* __restrict__ Kb,
    const unsigned short* __restrict__ Vt, const int* __restrict__ amask,
    const float* __restrict__ gates, unsigned short* __restrict__ Ctx,
    float* __restrict__ acc) {
  __shared__ unsigned short p_lds[8][16 * 32];     // per-wave P tile (bf16)
  __shared__ float s_red[4];

  const int lane = threadIdx.x & 31;
  const int wave = threadIdx.x >> 5;
  const int h = blockIdx.y, b = blockIdx.z;
  const int qbase = blockIdx.x * 128 + wave * 16;
  const int bh = b * NH_ + h;
  const unsigned short* Qh = Qb + (size_t)bh * S_LEN * HD_;
  const unsigned short* Kh = Kb + (size_t)bh * S_LEN * HD_;
  const unsigned short* Vth = Vt + (size_t)bh * HD_ * S_LEN;
  const int col = lane & 15;
  const int grp = (lane & 16) ? 8 : 0;
  const int kgrp = grp;
  const float SCALE = 0.125f;                      // 1/sqrt(64)

  if (threadIdx.x < 4) s_red[threadIdx.x] = 0.0f;
  __syncthreads();

  // Q A-fragments (d = 0..31, 32..63)
  const unsigned short* qrowp = Qh + (size_t)(qbase + col) * HD_;
  v16bf aq0 = load_frag_bf16(qrowp, kgrp);
  v16bf aq1 = load_frag_bf16(qrowp + 32, kgrp);

  // online-softmax state: slot i <-> row q = qbase + i + grp (replicated across 16 lanes)
  float m_i[8], l_i[8], ss[8], sd[8], sd2[8], tk[8][8];
#pragma unroll
  for (int i = 0; i < 8; i++) {
    m_i[i] = NEGINF; l_i[i] = 0.f; ss[i] = 0.f; sd[i] = 0.f; sd2[i] = 0.f;
#pragma unroll
    for (int e = 0; e < 8; e++) tk[i][e] = NEGINF;
  }
  v8f o0, o1, o2, o3;
#pragma unroll
  for (int i = 0; i < 8; i++) { o0[i] = 0.f; o1[i] = 0.f; o2[i] = 0.f; o3[i] = 0.f; }

  const int nkt = (qbase + 16 + 31) / 32;          // causal: keys <= qbase+15 < S
  for (int kt = 0; kt < nkt; ++kt) {
    const int kb = kt * 32;

    if (kt + 1 < nkt) {                            // prefetch next K/V tile
      __builtin_prefetch(Kh + (size_t)(kb + 32 + lane) * HD_, 0, 1);
      __builtin_prefetch(Vth + (size_t)lane * S_LEN + kb + 32, 0, 1);
      __builtin_prefetch(Vth + (size_t)(lane + 32) * S_LEN + kb + 32, 0, 1);
    }

    // K as B-fragments for QK^T: B[k=d][n=key]; two b128 loads each
    const int key0 = kb + col, key1 = kb + 16 + col;
    const unsigned short* k0p = Kh + (size_t)key0 * HD_;
    const unsigned short* k1p = Kh + (size_t)key1 * HD_;
    v16bf bk00 = load_frag_bf16(k0p, kgrp);
    v16bf bk01 = load_frag_bf16(k0p + 32, kgrp);
    v16bf bk10 = load_frag_bf16(k1p, kgrp);
    v16bf bk11 = load_frag_bf16(k1p + 32, kgrp);

    v8f sc0, sc1;
#pragma unroll
    for (int i = 0; i < 8; i++) { sc0[i] = 0.f; sc1[i] = 0.f; }
    sc0 = wmma_bf16(aq0, bk00, sc0);
    sc0 = wmma_bf16(aq1, bk01, sc0);
    sc1 = wmma_bf16(aq0, bk10, sc1);
    sc1 = wmma_bf16(aq1, bk11, sc1);

    const int pv0 = amask[b * S_LEN + key0];
    const int pv1 = amask[b * S_LEN + key1];

    float al[8], p0v[8], p1v[8];
#pragma unroll
    for (int i = 0; i < 8; i++) {
      int q = qbase + i + grp;
      float s0 = sc0[i] * SCALE, s1 = sc1[i] * SCALE;
      float sm0 = (pv0 && key0 <= q) ? s0 : NEGINF;
      float sm1 = (pv1 && key1 <= q) ? s1 : NEGINF;

      // per-lane top-8 maintenance (constant indices -> stays in VGPRs)
#pragma unroll
      for (int two = 0; two < 2; two++) {
        float sv = two ? sm1 : sm0;
        float mv = tk[i][0];
#pragma unroll
        for (int e = 1; e < 8; e++) mv = fminf(mv, tk[i][e]);
        if (sv > mv) {
          bool done = false;
#pragma unroll
          for (int e = 0; e < 8; e++) {
            bool hit = !done && (tk[i][e] == mv);
            if (hit) { tk[i][e] = sv; done = true; }
          }
        }
      }

      float rm = rmax16(fmaxf(sm0, sm1));
      float mn = fmaxf(m_i[i], rm);
      float a = __expf(m_i[i] - mn);
      float p0 = __expf(sm0 - mn);
      float p1 = __expf(sm1 - mn);
      float d0 = fabsf((float)(q - key0)), d1 = fabsf((float)(q - key1));
      // finite s0/s1 in products (p==0 kills masked terms)
      l_i[i] = l_i[i] * a + rsum16(p0 + p1);
      ss[i]  = ss[i]  * a + rsum16(p0 * s0 + p1 * s1);
      sd[i]  = sd[i]  * a + rsum16(p0 * d0 + p1 * d1);
      sd2[i] = sd2[i] * a + rsum16(p0 * d0 * d0 + p1 * d1 * d1);
      m_i[i] = mn;
      al[i] = a; p0v[i] = p0; p1v[i] = p1;
    }

#pragma unroll
    for (int i = 0; i < 8; i++) { o0[i] *= al[i]; o1[i] *= al[i]; o2[i] *= al[i]; o3[i] *= al[i]; }

    // P: C-layout -> LDS -> A-layout (per-wave region; same-wave DS ordering)
#pragma unroll
    for (int i = 0; i < 8; i++) {
      int row = i + grp;
      p_lds[wave][row * 32 + col]      = f2bfbits(p0v[i]);
      p_lds[wave][row * 32 + 16 + col] = f2bfbits(p1v[i]);
    }
    __builtin_amdgcn_wave_barrier();
    v16bf pa;
    {
      Vb16 u;
      u.h[0] = *(const us8*)&p_lds[wave][col * 32 + kgrp];
      u.h[1] = *(const us8*)&p_lds[wave][col * 32 + 16 + kgrp];
      pa = u.v;
    }

    // O_chunk += P(16x32) @ V_chunk(32x16) ; V^T rows give contiguous key runs
#pragma unroll
    for (int dc = 0; dc < 4; dc++) {
      const unsigned short* vrow = Vth + (size_t)(dc * 16 + col) * S_LEN + kb;
      v16bf bv = load_frag_bf16(vrow, kgrp);
      if (dc == 0)      o0 = wmma_bf16(pa, bv, o0);
      else if (dc == 1) o1 = wmma_bf16(pa, bv, o1);
      else if (dc == 2) o2 = wmma_bf16(pa, bv, o2);
      else              o3 = wmma_bf16(pa, bv, o3);
    }
  }

  // epilogue: context write (gate folded in) + per-row descriptors
  const float gate = gates[h];
  float ent_acc = 0.f, md_acc = 0.f, msq_acc = 0.f, tk_acc = 0.f;
#pragma unroll
  for (int i = 0; i < 8; i++) {
    float inv = 1.0f / l_i[i];
    int q = qbase + i + grp;
    size_t base = ((size_t)(b * S_LEN + q)) * HID + h * HD_;
    Ctx[base + 0 * 16 + col] = f2bfbits(o0[i] * inv * gate);
    Ctx[base + 1 * 16 + col] = f2bfbits(o1[i] * inv * gate);
    Ctx[base + 2 * 16 + col] = f2bfbits(o2[i] * inv * gate);
    Ctx[base + 3 * 16 + col] = f2bfbits(o3[i] * inv * gate);

    float ent = (m_i[i] + logf(l_i[i])) - ss[i] * inv;
    float md = sd[i] * inv;
    float msq = sd2[i] * inv;

    // merge 16 per-lane top-8 lists -> row top-8 mass
    float mass = 0.f;
    for (int t = 0; t < 8; t++) {
      float lm = tk[i][0];
#pragma unroll
      for (int e = 1; e < 8; e++) lm = fmaxf(lm, tk[i][e]);
      float gm = rmax16(lm);
      mass += __expf(gm - m_i[i]);
      bool rem = false;
#pragma unroll
      for (int e = 0; e < 8; e++) {
        bool hit = !rem && (tk[i][e] == gm);
        if (hit) { tk[i][e] = NEGINF; rem = true; }
      }
    }
    ent_acc += ent; md_acc += md; msq_acc += msq; tk_acc += mass * inv;
  }

  if (col == 0) {  // lanes 0 and 16 hold distinct row groups; stats replicated otherwise
    atomicAdd(&s_red[0], ent_acc);
    atomicAdd(&s_red[1], md_acc);
    atomicAdd(&s_red[2], msq_acc);
    atomicAdd(&s_red[3], tk_acc);
  }
  __syncthreads();
  if (threadIdx.x == 0) {
    atomicAdd(&acc[h * 4 + 0], s_red[0]);
    atomicAdd(&acc[h * 4 + 1], s_red[1]);
    atomicAdd(&acc[h * 4 + 2], s_red[2]);
    atomicAdd(&acc[h * 4 + 3], s_red[3]);
  }
}

// ---------------------------------------------------------------------------
// Kernel 3: out = ctx @ Wout + bout  (M=4096, N=1024, K=1024)
// ---------------------------------------------------------------------------
__global__ void __launch_bounds__(256) out_gemm_kernel(
    const unsigned short* __restrict__ Ctx, const unsigned short* __restrict__ WoutT,
    const float* __restrict__ bout, float* __restrict__ out) {
  const int lane = threadIdx.x & 31;
  const int wave = threadIdx.x >> 5;
  const int tile = blockIdx.x * 8 + wave;          // 256*64 tiles
  const int NT_N = HID / 16;                       // 64
  const int tm = tile / NT_N, tn = tile % NT_N;
  const int col = lane & 15;
  const int grp = (lane & 16) ? 8 : 0;
  const int kgrp = grp;

  v8f acc;
#pragma unroll
  for (int i = 0; i < 8; i++) acc[i] = 0.0f;

  const unsigned short* crow = Ctx + (size_t)(tm * 16 + col) * HID;
  const unsigned short* wrow = WoutT + (size_t)(tn * 16 + col) * HID;
  for (int kk = 0; kk < HID; kk += 32) {
    if (kk + 32 < HID) {
      __builtin_prefetch(crow + kk + 32, 0, 1);
      __builtin_prefetch(wrow + kk + 32, 0, 1);
    }
    v16bf ua = load_frag_bf16(crow + kk, kgrp);
    v16bf ub = load_frag_bf16(wrow + kk, kgrp);
    acc = wmma_bf16(ua, ub, acc);
  }

  const int n = tn * 16 + col;
  const float bias = bout[n];
#pragma unroll
  for (int i = 0; i < 8; i++) {
    int m = tm * 16 + i + grp;
    out[(size_t)m * HID + n] = acc[i] + bias;
  }
}

// ---------------------------------------------------------------------------
// Kernel 4: finalize [NH,4] descriptors
// ---------------------------------------------------------------------------
__global__ void finalize_kernel(const float* __restrict__ acc, float* __restrict__ desc) {
  int h = threadIdx.x;
  if (h >= NH_) return;
  const float inv = 1.0f / (float)(BATCH * S_LEN);
  float ent = acc[h * 4 + 0] * inv;
  float md  = acc[h * 4 + 1] * inv;
  float msq = acc[h * 4 + 2] * inv;
  float tkm = acc[h * 4 + 3] * inv;
  desc[h * 4 + 0] = ent;
  desc[h * 4 + 1] = md;
  desc[h * 4 + 2] = sqrtf(fmaxf(msq - md * md, 0.0f));
  desc[h * 4 + 3] = tkm;
}

// ---------------------------------------------------------------------------
extern "C" void kernel_launch(void* const* d_in, const int* in_sizes, int n_in,
                              void* d_out, int out_size, void* d_ws, size_t ws_size,
                              hipStream_t stream) {
  (void)in_sizes; (void)n_in; (void)out_size; (void)ws_size;
  const float* x     = (const float*)d_in[0];
  const int*   amask = (const int*)d_in[1];
  const float* gates = (const float*)d_in[2];
  const float* Wqkv  = (const float*)d_in[3];
  const float* bqkv  = (const float*)d_in[4];
  const float* Wout  = (const float*)d_in[5];
  const float* bout  = (const float*)d_in[6];
  float* out = (float*)d_out;

  const size_t PHE   = (size_t)BATCH * NH_ * S_LEN * HD_;  // 4,194,304
  const size_t NX    = (size_t)BATCH * S_LEN * HID;        // 4,194,304
  unsigned short* Qb    = (unsigned short*)d_ws;
  unsigned short* Kb    = Qb + PHE;
  unsigned short* Vt    = Kb + PHE;                        // [B,NH,64,S]
  unsigned short* Ctx   = Vt + PHE;
  unsigned short* xbuf  = Ctx + NX;
  unsigned short* WqkvT = xbuf + NX;                       // [3072,1024]
  unsigned short* WoutT = WqkvT + (size_t)3 * HID * HID;   // [1024,1024]
  float* acc = (float*)(WoutT + (size_t)HID * HID);

  hipLaunchKernelGGL(init_acc_kernel, dim3(1), dim3(64), 0, stream, acc);
  hipLaunchKernelGGL(pack_x_kernel, dim3(NX / 8 / 256), dim3(256), 0, stream, x, xbuf);
  hipLaunchKernelGGL(packT_kernel, dim3((3 * HID) / 32, HID / 32), dim3(256), 0, stream,
                     Wqkv, WqkvT, HID, 3 * HID);
  hipLaunchKernelGGL(packT_kernel, dim3(HID / 32, HID / 32), dim3(256), 0, stream,
                     Wout, WoutT, HID, HID);
  hipLaunchKernelGGL(qkv_gemm_kernel, dim3((256 * 192) / 8), dim3(256), 0, stream,
                     xbuf, WqkvT, bqkv, Qb, Kb, Vt);
  hipLaunchKernelGGL(attn_kernel, dim3(S_LEN / 128, NH_, BATCH), dim3(256), 0, stream,
                     Qb, Kb, Vt, amask, gates, Ctx, acc);
  hipLaunchKernelGGL(out_gemm_kernel, dim3((256 * 64) / 8), dim3(256), 0, stream,
                     Ctx, WoutT, bout, out);
  hipLaunchKernelGGL(finalize_kernel, dim3(1), dim3(32), 0, stream,
                     acc, out + NX);
}
